// EdgeBlock_17729624998201
// MI455X (gfx1250) — matrix operand order repaired and checked by reading, not compiled, not run on hardware
//
#include <hip/hip_runtime.h>

typedef __attribute__((ext_vector_type(16))) _Float16 v16h;
typedef __attribute__((ext_vector_type(8)))  _Float16 v8h;
typedef __attribute__((ext_vector_type(8)))  float    v8f;
typedef __attribute__((ext_vector_type(4)))  float    f4;

#define WAVES_PER_BLOCK 8

// out[16 edges, 64] = relu(X[16,224] @ W1[224,32] + b1) @ W2[32,64] + b2
// X = concat(edge_attr[e], node_attr[snd], node_attr[rcv], global_attr)
__global__ __launch_bounds__(256) void edge_block_kernel(
    const float* __restrict__ node_attr,
    const float* __restrict__ edge_attr,
    const float* __restrict__ global_attr,
    const float* __restrict__ W1,
    const float* __restrict__ b1,
    const float* __restrict__ W2,
    const float* __restrict__ b2,
    const int*   __restrict__ edge_index,
    float* __restrict__ out,
    int E, int ntiles)
{
    // Transposed f16 weights in LDS: B-fragments become contiguous ds_load_b128 pairs.
    __shared__ __align__(32) _Float16 w1t[32 * 224];          // [n][k]
    __shared__ __align__(32) _Float16 w2t[64 * 32];           // [n][k]
    __shared__ float b1s[32];
    __shared__ float b2s[64];
    __shared__ __align__(32) _Float16 hbuf[WAVES_PER_BLOCK][16 * 32]; // per-wave h tile

    for (int i = threadIdx.x; i < 224 * 32; i += 256) {
        int k = i >> 5, n = i & 31;
        w1t[n * 224 + k] = (_Float16)W1[i];
    }
    for (int i = threadIdx.x; i < 32 * 64; i += 256) {
        int k = i >> 6, n = i & 63;
        w2t[n * 32 + k] = (_Float16)W2[i];
    }
    if (threadIdx.x < 32) b1s[threadIdx.x] = b1[threadIdx.x];
    if (threadIdx.x < 64) b2s[threadIdx.x] = b2[threadIdx.x];
    __syncthreads();

    const int lane  = threadIdx.x & 31;
    const int wave  = threadIdx.x >> 5;
    const int row16 = lane & 15;   // M row (A/C) or N column (B/C)
    const int half  = lane >> 4;   // lane group selects K sub-range
    _Float16* hl = &hbuf[wave][0];

    for (int tile = blockIdx.x * WAVES_PER_BLOCK + wave; tile < ntiles;
         tile += gridDim.x * WAVES_PER_BLOCK) {
        const int e0 = tile * 16;
        int e = e0 + row16;
        if (e >= E) e = E - 1;  // clamp tail: keep EXEC all-ones for WMMA
        const int snd = __builtin_nontemporal_load(&edge_index[e]);
        const int rcv = __builtin_nontemporal_load(&edge_index[E + e]);

        // GEMM1 accumulators, bias-initialized (C[m][n] = b1[n])
        v8f c0, c1;
        {
            const float bv0 = b1s[row16];
            const float bv1 = b1s[16 + row16];
            #pragma unroll
            for (int r = 0; r < 8; ++r) { c0[r] = bv0; c1[r] = bv1; }
        }

        #pragma unroll
        for (int kb = 0; kb < 7; ++kb) {
            const float* base;
            bool streaming;
            if (kb < 2)      { base = edge_attr + (size_t)e   * 64 + kb * 32;      streaming = true;  }
            else if (kb < 4) { base = node_attr + (size_t)snd * 64 + (kb - 2) * 32; streaming = false; }
            else if (kb < 6) { base = node_attr + (size_t)rcv * 64 + (kb - 4) * 32; streaming = false; }
            else             { base = global_attr;                                  streaming = false; }

            // A-frag k order chosen here (same as B, not AMD table order): lanes 0-15 hold K=[0..15],
            // lanes 16-31 hold K=[16..31] -> two contiguous 8-float runs.
            const f4* p1 = (const f4*)(base + half * 16);
            const f4* p2 = (const f4*)(base + half * 16 + 8);
            f4 f0, f1, g0, g1;
            if (streaming) {
                f0 = __builtin_nontemporal_load(p1);
                f1 = __builtin_nontemporal_load(p1 + 1);
                g0 = __builtin_nontemporal_load(p2);
                g1 = __builtin_nontemporal_load(p2 + 1);
            } else {  // node_attr/global: let it stay resident in 192MB L2
                f0 = p1[0]; f1 = p1[1]; g0 = p2[0]; g1 = p2[1];
            }
            v16h a;
            #pragma unroll
            for (int t = 0; t < 4; ++t) {
                a[t]      = (_Float16)f0[t];
                a[4 + t]  = (_Float16)f1[t];
                a[8 + t]  = (_Float16)g0[t];
                a[12 + t] = (_Float16)g1[t];
            }

            // B-frag: lane holds column row16(+16); K = half*16 .. +15 contiguous.
            const v16h b0 = *(const v16h*)&w1t[(row16)      * 224 + kb * 32 + half * 16];
            const v16h b1f = *(const v16h*)&w1t[(16 + row16) * 224 + kb * 32 + half * 16];

            c0 = __builtin_amdgcn_wmma_f32_16x16x32_f16(false, a, false, b0,  (short)0, c0, false, false);
            c1 = __builtin_amdgcn_wmma_f32_16x16x32_f16(false, a, false, b1f, (short)0, c1, false, false);
        }

        // ReLU + f16, remap C layout -> row-major h tile in LDS
        #pragma unroll
        for (int r = 0; r < 8; ++r) {
            const int m = r + 8 * half;
            float v0 = c0[r] > 0.f ? c0[r] : 0.f;
            float v1 = c1[r] > 0.f ? c1[r] : 0.f;
            hl[m * 32 + row16]      = (_Float16)v0;
            hl[m * 32 + 16 + row16] = (_Float16)v1;
        }
        asm volatile("s_wait_dscnt 0" ::: "memory");

        // GEMM2: A = h[16,32] from LDS in A-fragment layout
        v8h r1 = *(const v8h*)&hl[row16 * 32 + half * 16];
        v8h r2 = *(const v8h*)&hl[row16 * 32 + half * 16 + 8];
        v16h ha;
        #pragma unroll
        for (int t = 0; t < 8; ++t) { ha[t] = r1[t]; ha[8 + t] = r2[t]; }

        v8f d0, d1, d2, d3;
        {
            const float bv0 = b2s[row16];
            const float bv1 = b2s[16 + row16];
            const float bv2 = b2s[32 + row16];
            const float bv3 = b2s[48 + row16];
            #pragma unroll
            for (int r = 0; r < 8; ++r) { d0[r] = bv0; d1[r] = bv1; d2[r] = bv2; d3[r] = bv3; }
        }
        const v16h wb0 = *(const v16h*)&w2t[(row16)      * 32 + half * 16];
        const v16h wb1 = *(const v16h*)&w2t[(16 + row16) * 32 + half * 16];
        const v16h wb2 = *(const v16h*)&w2t[(32 + row16) * 32 + half * 16];
        const v16h wb3 = *(const v16h*)&w2t[(48 + row16) * 32 + half * 16];

        d0 = __builtin_amdgcn_wmma_f32_16x16x32_f16(false, ha, false, wb0, (short)0, d0, false, false);
        d1 = __builtin_amdgcn_wmma_f32_16x16x32_f16(false, ha, false, wb1, (short)0, d1, false, false);
        d2 = __builtin_amdgcn_wmma_f32_16x16x32_f16(false, ha, false, wb2, (short)0, d2, false, false);
        d3 = __builtin_amdgcn_wmma_f32_16x16x32_f16(false, ha, false, wb3, (short)0, d3, false, false);

        // Store D tiles: lane l, VGPR r -> row (r + 8*half), col (nt*16 + row16)
        #pragma unroll
        for (int r = 0; r < 8; ++r) {
            const int m = e0 + r + 8 * half;
            if (m < E) {
                float* po = out + (size_t)m * 64 + row16;
                __builtin_nontemporal_store(d0[r], po);
                __builtin_nontemporal_store(d1[r], po + 16);
                __builtin_nontemporal_store(d2[r], po + 32);
                __builtin_nontemporal_store(d3[r], po + 48);
            }
        }
    }
}

extern "C" void kernel_launch(void* const* d_in, const int* in_sizes, int n_in,
                              void* d_out, int out_size, void* d_ws, size_t ws_size,
                              hipStream_t stream) {
    const float* node_attr   = (const float*)d_in[0];
    const float* edge_attr   = (const float*)d_in[1];
    const float* global_attr = (const float*)d_in[2];
    const float* W1          = (const float*)d_in[3];
    const float* b1          = (const float*)d_in[4];
    const float* W2          = (const float*)d_in[5];
    const float* b2          = (const float*)d_in[6];
    const int*   edge_index  = (const int*)d_in[7];
    float*       out         = (float*)d_out;

    const int E = in_sizes[1] / 64;
    const int ntiles = (E + 15) / 16;
    int blocks = (ntiles + WAVES_PER_BLOCK - 1) / WAVES_PER_BLOCK;
    if (blocks > 1024) blocks = 1024;
    if (blocks < 1) blocks = 1;

    edge_block_kernel<<<blocks, 256, 0, stream>>>(
        node_attr, edge_attr, global_attr, W1, b1, W2, b2, edge_index,
        out, E, ntiles);
}
